// MixtureOfExperts_3710851744201
// MI455X (gfx1250) — matrix operand order repaired and checked by reading, not compiled, not run on hardware
//
#include <hip/hip_runtime.h>
#include <cmath>

#define D_MODEL 1024
#define D_FF    2048
#define N_EXP   8
#define TM      64        // tokens per tile
#define WAVE    32
#define NTHREADS 512      // 16 waves: 4 (M) x 4 (N)
#define FFCHUNK 128
#define XS_LD   (D_MODEL + 8)   // padded LDS row stride (halfwords)
#define HS_LD   (FFCHUNK + 8)

typedef __attribute__((ext_vector_type(16))) __bf16    v16bf;
typedef __attribute__((ext_vector_type(8)))  float     v8f;
typedef __attribute__((ext_vector_type(4)))  unsigned  u32x4;
typedef __attribute__((ext_vector_type(2)))  unsigned  u32x2;
typedef __attribute__((ext_vector_type(4)))  float     f32x4;

union BFrag {
    v16bf    v;
    u32x4    q[2];
    unsigned u[8];
};

__device__ __forceinline__ unsigned short f2bf(float f) {
    unsigned u = __float_as_uint(f);
    u += 0x7fffu + ((u >> 16) & 1u);          // round-to-nearest-even
    return (unsigned short)(u >> 16);
}
__device__ __forceinline__ unsigned pack2(float a, float b) {
    return (unsigned)f2bf(a) | ((unsigned)f2bf(b) << 16);
}

// ---------------------------------------------------------------- router
__global__ void router_kernel(const float* __restrict__ x,
                              const float* __restrict__ gate_w,
                              int2* __restrict__ r_idx,
                              float2* __restrict__ r_w, int T) {
    int t = blockIdx.x * blockDim.x + threadIdx.x;
    if (t >= T) return;
    const float* xp = x + (size_t)t * D_MODEL;
    float acc[N_EXP];
#pragma unroll
    for (int e = 0; e < N_EXP; ++e) acc[e] = 0.f;
    for (int d = 0; d < D_MODEL; ++d) {
        float xv = xp[d];
        const float* g = gate_w + d * N_EXP;
#pragma unroll
        for (int e = 0; e < N_EXP; ++e) acc[e] = fmaf(xv, g[e], acc[e]);
    }
    int i0 = 0; float v0 = acc[0];
#pragma unroll
    for (int e = 1; e < N_EXP; ++e) if (acc[e] > v0) { v0 = acc[e]; i0 = e; }
    int i1 = -1; float v1 = -INFINITY;
#pragma unroll
    for (int e = 0; e < N_EXP; ++e) if (e != i0 && acc[e] > v1) { v1 = acc[e]; i1 = e; }
    float e1 = expf(v1 - v0);
    float w0 = 1.f / (1.f + e1);
    r_idx[t] = make_int2(i0, i1);
    r_w[t]   = make_float2(w0, e1 * w0);
}

// ----------------------------------------------- stable per-expert compaction
__global__ void build_lists(const int2* __restrict__ r_idx,
                            const float2* __restrict__ r_w,
                            int* __restrict__ lists, float* __restrict__ wts,
                            int* __restrict__ counts, int T) {
    const int e = blockIdx.x;
    const int lane = threadIdx.x;           // 32 threads = 1 wave
    int base = 0;
    for (int t0 = 0; t0 < T; t0 += WAVE) {
        int t = t0 + lane;
        int2   id = r_idx[t];
        float2 w  = r_w[t];
        bool m0 = (id.x == e), m1 = (id.y == e);
        bool m  = m0 | m1;
        float wgt = m0 ? w.x : w.y;
        unsigned long long bm = __ballot(m);
        int pos = base + __popcll(bm & ((1ull << lane) - 1ull));
        if (m) { lists[e * T + pos] = t; wts[e * T + pos] = wgt; }
        base += __popcll(bm);
    }
    if (lane == 0) counts[e] = base;
}

// ------------------------------------------------ fp32 -> bf16 weight convert
__global__ __launch_bounds__(256)
void cvt_bf16_kernel(const float* __restrict__ src,
                     unsigned short* __restrict__ dst, size_t n4) {
    size_t stride = (size_t)gridDim.x * blockDim.x;
    for (size_t i = (size_t)blockIdx.x * blockDim.x + threadIdx.x; i < n4; i += stride) {
        f32x4 f = reinterpret_cast<const f32x4*>(src)[i];
        u32x2 o; o.x = pack2(f.x, f.y); o.y = pack2(f.z, f.w);
        reinterpret_cast<u32x2*>(dst)[i] = o;
    }
}

// B-fragment loader: one column of the 32x16 K-major tile per lane, K in ISA table order
template <bool WB16>
__device__ __forceinline__ BFrag loadB(const float* __restrict__ wf,
                                       const unsigned short* __restrict__ wb,
                                       size_t row, int col, int ld) {
    BFrag b;
    const int kh = (threadIdx.x & 16) >> 1;
#pragma unroll
    for (int i = 0; i < 8; ++i) {
        size_t k0 = row + (i < 4 ? kh + 2 * i : 16 + kh + 2 * (i - 4));
        if (WB16) {
            b.u[i] = (unsigned)wb[k0 * ld + col] | ((unsigned)wb[(k0 + 1) * ld + col] << 16);
        } else {
            b.u[i] = pack2(wf[k0 * ld + col], wf[(k0 + 1) * ld + col]);
        }
    }
    return b;
}

// ---------------------------------------------------- fused expert MLP (WMMA)
template <bool WB16>
__global__ __launch_bounds__(NTHREADS)
void moe_mlp_kernel(const float* __restrict__ x,
                    const float* __restrict__ w1f,
                    const float* __restrict__ w2f,
                    const unsigned short* __restrict__ w1b,
                    const unsigned short* __restrict__ w2b,
                    const int* __restrict__ lists,
                    const float* __restrict__ wts,
                    const int* __restrict__ counts,
                    float* __restrict__ out, int T) {
    const int e    = blockIdx.y;
    const int tile = blockIdx.x;
    const int cnt  = counts[e];
    if (tile * TM >= cnt) return;

    const int tid   = threadIdx.x;
    const int lane  = tid & (WAVE - 1);
    const int wv    = tid >> 5;             // 16 waves
    const int wm    = wv & 3;               // M tile (16 rows each)
    const int wn    = wv >> 2;              // N group
    const int r16   = lane & 15;
    const int khalf = (lane >> 4) * 8;      // A-fragment K sub-offset (ISA layout)
    const int crow  = (lane >> 4) << 3;     // C/D row offset (ISA layout)

    __shared__ __align__(16) unsigned short xs[TM * XS_LD];   // ~132 KB bf16 x-tile
    __shared__ __align__(16) unsigned short hs[TM * HS_LD];   // ~17 KB bf16 h-chunk
    __shared__ int   s_tok[TM];
    __shared__ float s_wt[TM];

    if (tid < TM) {
        int slot = tile * TM + tid;
        if (slot < cnt) { s_tok[tid] = lists[e * T + slot]; s_wt[tid] = wts[e * T + slot]; }
        else            { s_tok[tid] = 0;                   s_wt[tid] = 0.f; }
    }
    __syncthreads();

    // stage x tile -> LDS bf16 (padded rows)
    for (int i = tid; i < TM * D_MODEL / 4; i += NTHREADS) {
        int row = i >> 8;                   // 256 float4 per row
        int c4  = i & 255;
        f32x4 f = reinterpret_cast<const f32x4*>(x + (size_t)s_tok[row] * D_MODEL)[c4];
        unsigned* dst = reinterpret_cast<unsigned*>(&xs[row * XS_LD + c4 * 4]);
        dst[0] = pack2(f.x, f.y);
        dst[1] = pack2(f.z, f.w);
    }
    __syncthreads();

    const float*          w1fe = w1f + (size_t)e * D_MODEL * D_FF;
    const float*          w2fe = w2f + (size_t)e * D_FF * D_MODEL;
    const unsigned short* w1be = w1b + (size_t)e * D_MODEL * D_FF;
    const unsigned short* w2be = w2b + (size_t)e * D_FF * D_MODEL;

    v8f yacc[16];                           // wave owns 16 rows x 256 cols
#pragma unroll
    for (int n = 0; n < 16; ++n)
#pragma unroll
        for (int j = 0; j < 8; ++j) yacc[n][j] = 0.f;

    const unsigned short* xrow = &xs[(wm * 16 + r16) * XS_LD + khalf];

    for (int fc = 0; fc < D_FF / FFCHUNK; ++fc) {
        // -------- h(16x32 per wave) = silu(x @ W1), K = 1024
        v8f hacc[2];
#pragma unroll
        for (int c = 0; c < 2; ++c)
#pragma unroll
            for (int j = 0; j < 8; ++j) hacc[c][j] = 0.f;
        const int n1 = fc * FFCHUNK + wn * 16;
#pragma unroll 4
        for (int kk = 0; kk < D_MODEL / 32; ++kk) {
            const int kb = kk * 32;
            BFrag a;
            a.q[0] = *reinterpret_cast<const u32x4*>(xrow + kb);
            a.q[1] = *reinterpret_cast<const u32x4*>(xrow + kb + 16);
#pragma unroll
            for (int c = 0; c < 2; ++c) {
                BFrag b = loadB<WB16>(w1fe, w1be, (size_t)kb, n1 + c * 64 + r16, D_FF);
                hacc[c] = __builtin_amdgcn_wmma_f32_16x16x32_bf16(
                    false, a.v, false, b.v, (short)0, hacc[c], false, false);
            }
        }
        // silu + store bf16 to LDS
#pragma unroll
        for (int c = 0; c < 2; ++c)
#pragma unroll
            for (int j = 0; j < 8; ++j) {
                float v = hacc[c][j];
                float s = v / (1.f + expf(-v));
                hs[(wm * 16 + j + crow) * HS_LD + wn * 16 + c * 64 + r16] = f2bf(s);
            }
        __syncthreads();

        // -------- y(16x256 per wave) += h @ W2, K = 128
        const unsigned short* hrow = &hs[(wm * 16 + r16) * HS_LD + khalf];
#pragma unroll
        for (int kk = 0; kk < FFCHUNK / 32; ++kk) {
            const int kb = kk * 32;
            BFrag a;
            a.q[0] = *reinterpret_cast<const u32x4*>(hrow + kb);
            a.q[1] = *reinterpret_cast<const u32x4*>(hrow + kb + 16);
            const size_t krow = (size_t)(fc * FFCHUNK + kb);
#pragma unroll
            for (int nt = 0; nt < 16; ++nt) {
                BFrag b = loadB<WB16>(w2fe, w2be, krow, wn * 256 + nt * 16 + r16, D_MODEL);
                yacc[nt] = __builtin_amdgcn_wmma_f32_16x16x32_bf16(
                    false, a.v, false, b.v, (short)0, yacc[nt], false, false);
            }
        }
        __syncthreads();
    }

    // -------- weighted scatter: per out element, its TOP_K contributions (+ exact zeros)
#pragma unroll
    for (int nt = 0; nt < 16; ++nt) {
#pragma unroll
        for (int j = 0; j < 8; ++j) {
            int row = wm * 16 + j + crow;
            int col = wn * 256 + nt * 16 + r16;
            float v = yacc[nt][j] * s_wt[row];
            atomicAdd(&out[(size_t)s_tok[row] * D_MODEL + col], v);
        }
    }
}

// ---------------------------------------------------------------- launcher
extern "C" void kernel_launch(void* const* d_in, const int* in_sizes, int n_in,
                              void* d_out, int out_size, void* d_ws, size_t ws_size,
                              hipStream_t stream) {
    const float* x      = (const float*)d_in[0];
    const float* gate_w = (const float*)d_in[1];
    const float* w1     = (const float*)d_in[2];
    const float* w2     = (const float*)d_in[3];
    float* out = (float*)d_out;
    const int T = in_sizes[0] / D_MODEL;     // B*S = 4096

    char* p0 = (char*)d_ws;
    char* p  = p0;
    int2*   r_idx  = (int2*)p;   p += (size_t)T * sizeof(int2);
    float2* r_w    = (float2*)p; p += (size_t)T * sizeof(float2);
    int*    lists  = (int*)p;    p += (size_t)N_EXP * T * sizeof(int);
    float*  wts    = (float*)p;  p += (size_t)N_EXP * T * sizeof(float);
    int*    counts = (int*)p;    p += 256;   // counts + pad
    p = (char*)(((uintptr_t)p + 255) & ~(uintptr_t)255);

    const size_t welems = (size_t)N_EXP * D_MODEL * D_FF;  // per weight tensor
    unsigned short* w1b = (unsigned short*)p;
    unsigned short* w2b = w1b + welems;
    const size_t need = (size_t)((char*)(w2b + welems) - p0);
    const bool use_bf16w = (ws_size >= need);

    hipMemsetAsync(d_out, 0, (size_t)out_size * sizeof(float), stream);

    router_kernel<<<(T + 255) / 256, 256, 0, stream>>>(x, gate_w, r_idx, r_w, T);
    build_lists<<<N_EXP, WAVE, 0, stream>>>(r_idx, r_w, lists, wts, counts, T);

    dim3 grid((T + TM - 1) / TM, N_EXP);
    if (use_bf16w) {
        cvt_bf16_kernel<<<4096, 256, 0, stream>>>(w1, w1b, welems / 4);
        cvt_bf16_kernel<<<4096, 256, 0, stream>>>(w2, w2b, welems / 4);
        moe_mlp_kernel<true><<<grid, NTHREADS, 0, stream>>>(
            x, w1, w2, w1b, w2b, lists, wts, counts, out, T);
    } else {
        moe_mlp_kernel<false><<<grid, NTHREADS, 0, stream>>>(
            x, w1, w2, w1b, w2b, lists, wts, counts, out, T);
    }
}
